// LDAM_44624710205470
// MI455X (gfx1250) — matrix operand order repaired and checked by reading, not compiled, not run on hardware
//
#include <hip/hip_runtime.h>
#include <hip/hip_bf16.h>

typedef __bf16 bf16_t;
typedef __attribute__((ext_vector_type(16))) __bf16 v16bf;
typedef __attribute__((ext_vector_type(8)))  float   v8f;

#define BATCH 8
#define CDIM  256
#define ICDIM 128
#define NQ    4096
#define NKV   1024

union BF16x16 {
  v16bf v;
  uint4 q[2];
  unsigned int u[8];
};
union BF16x8 {
  bf16_t h[8];
  uint4 q;
};

__device__ __forceinline__ v8f v8zero() {
  v8f z = {0.f,0.f,0.f,0.f,0.f,0.f,0.f,0.f};
  return z;
}

__device__ __forceinline__ v8f wmma_bf16(v16bf a, v16bf b, v8f c) {
  return __builtin_amdgcn_wmma_f32_16x16x32_bf16(false, a, false, b, (short)0, c, false, false);
}

__device__ __forceinline__ unsigned int pk2(float lo, float hi) {
  union { bf16_t h[2]; unsigned int u; } p;
  p.h[0] = (bf16_t)lo; p.h[1] = (bf16_t)hi;
  return p.u;
}

// ---------------------------------------------------------------------------
// 2x2 maxpool of [B,256,128,128] -> f32 [b][c][4096] (optional) and
// transposed bf16 [b][q][256] (for GEMM B-operands), via LDS tile transpose.
// grid: (4 c-tiles of 64, 64 h, 8 b), block 256
// ---------------------------------------------------------------------------
template <bool WF32>
__global__ void k_pool_in(const float* __restrict__ x, float* __restrict__ f32out,
                          bf16_t* __restrict__ bfout) {
  __shared__ bf16_t lds[64][65];
  int b = blockIdx.z, h = blockIdx.y, c0 = blockIdx.x * 64;
  int t = threadIdx.x;
  const float* base = x + (((size_t)b * CDIM + c0) * 128 + 2 * h) * 128;
  for (int i = t; i < 4096; i += 256) {
    int cl = i >> 6, w = i & 63;
    const float* p = base + (size_t)cl * 16384 + 2 * w;
    float2 a = *(const float2*)(p);
    float2 c = *(const float2*)(p + 128);
    float m = fmaxf(fmaxf(a.x, a.y), fmaxf(c.x, c.y));
    if (WF32) f32out[((size_t)b * CDIM + c0 + cl) * NQ + h * 64 + w] = m;
    lds[cl][w] = (bf16_t)m;
  }
  __syncthreads();
  for (int i = t; i < 4096; i += 256) {
    int w = i >> 6, cl = i & 63;
    bfout[((size_t)b * NQ + h * 64 + w) * CDIM + c0 + cl] = lds[cl][w];
  }
}

// ---------------------------------------------------------------------------
// Convert all 4 weight matrices to bf16: theta(32768) g(32768) phi(32768) W(32768)
// ---------------------------------------------------------------------------
__global__ void k_cvt_w(const float* __restrict__ th, const float* __restrict__ g,
                        const float* __restrict__ ph, const float* __restrict__ Ww,
                        bf16_t* __restrict__ out) {
  int i = blockIdx.x * blockDim.x + threadIdx.x;
  float v;
  if (i < 32768) v = th[i];
  else if (i < 65536) v = g[i - 32768];
  else if (i < 98304) v = ph[i - 65536];
  else v = Ww[i - 98304];
  out[i] = (bf16_t)v;
}

// ---------------------------------------------------------------------------
// WMMA GEMM: out[b][o][q] (f32, channel-major) = W[o][c] * X[b][q][c] + bias[o]
// one wave per 16x16 tile. grid (NQ/16, O/16, B), block 32.
// ---------------------------------------------------------------------------
__global__ void __launch_bounds__(32) k_gemm_ct(const bf16_t* __restrict__ X,
                                                const bf16_t* __restrict__ Wt,
                                                const float* __restrict__ bias,
                                                float* __restrict__ out,
                                                int C, int O) {
  int l = threadIdx.x, lm = l & 15, hi = l >> 4;
  int qt = blockIdx.x * 16, ot = blockIdx.y * 16, b = blockIdx.z;
  const bf16_t* Xb = X + (size_t)b * NQ * C;
  v8f acc = v8zero();
  for (int k = 0; k < C; k += 32) {
    BF16x16 A, Bm;
    const bf16_t* wr = Wt + (size_t)(ot + lm) * C + k + hi * 16;
    A.q[0] = *(const uint4*)(wr);
    A.q[1] = *(const uint4*)(wr + 8);
    const bf16_t* xr = Xb + (size_t)(qt + lm) * C + k + hi * 16;
    Bm.q[0] = *(const uint4*)(xr);
    Bm.q[1] = *(const uint4*)(xr + 8);
    acc = wmma_bf16(A.v, Bm.v, acc);
  }
  int obase = ot + hi * 8;
#pragma unroll
  for (int j = 0; j < 8; ++j) {
    out[((size_t)b * O + obase + j) * NQ + qt + lm] = acc[j] + bias[obase + j];
  }
}

// ---------------------------------------------------------------------------
// Same GEMM but bf16 output in [b][q][O] layout (for theta). grid (NQ/16, O/16, B)
// ---------------------------------------------------------------------------
__global__ void __launch_bounds__(32) k_gemm_qt(const bf16_t* __restrict__ X,
                                                const bf16_t* __restrict__ Wt,
                                                const float* __restrict__ bias,
                                                bf16_t* __restrict__ out,
                                                int C, int O) {
  int l = threadIdx.x, lm = l & 15, hi = l >> 4;
  int qt = blockIdx.x * 16, ot = blockIdx.y * 16, b = blockIdx.z;
  const bf16_t* Xb = X + (size_t)b * NQ * C;
  v8f acc = v8zero();
  for (int k = 0; k < C; k += 32) {
    BF16x16 A, Bm;
    const bf16_t* wr = Wt + (size_t)(ot + lm) * C + k + hi * 16;
    A.q[0] = *(const uint4*)(wr);
    A.q[1] = *(const uint4*)(wr + 8);
    const bf16_t* xr = Xb + (size_t)(qt + lm) * C + k + hi * 16;
    Bm.q[0] = *(const uint4*)(xr);
    Bm.q[1] = *(const uint4*)(xr + 8);
    acc = wmma_bf16(A.v, Bm.v, acc);
  }
  int obase = ot + hi * 8;
  BF16x8 o8;
#pragma unroll
  for (int j = 0; j < 8; ++j) o8.h[j] = (bf16_t)(acc[j] + bias[obase + j]);
  *(uint4*)(out + ((size_t)b * NQ + qt + lm) * O + obase) = o8.q;
}

// ---------------------------------------------------------------------------
// g pool: gc f32 [b][ic][64][64] -> 2x2 max -> gT bf16 [b][ic][1024]
// ---------------------------------------------------------------------------
__global__ void k_pool_g(const float* __restrict__ gc, bf16_t* __restrict__ gt) {
  int i = blockIdx.x * 256 + threadIdx.x;  // < 8*128*1024
  int kv = i & 1023, ic = (i >> 10) & 127, b = i >> 17;
  int hh = kv >> 5, ww = kv & 31;
  const float* p = gc + ((size_t)(b * ICDIM + ic)) * NQ + hh * 128 + 2 * ww;
  float m = fmaxf(fmaxf(p[0], p[1]), fmaxf(p[64], p[65]));
  gt[i] = (bf16_t)m;
}

// ---------------------------------------------------------------------------
// phi pool + transpose: phic f32 [b][ic][64][64] -> phiT bf16 [b][kv][128]
// grid (32 hh, 8 b), block 256
// ---------------------------------------------------------------------------
__global__ void k_pool_phi(const float* __restrict__ pc, bf16_t* __restrict__ pt) {
  __shared__ bf16_t lds[128][33];
  int b = blockIdx.y, hh = blockIdx.x;
  for (int i = threadIdx.x; i < 4096; i += 256) {
    int ic = i >> 5, ww = i & 31;
    const float* p = pc + ((size_t)(b * ICDIM + ic)) * NQ + hh * 128 + 2 * ww;
    float m = fmaxf(fmaxf(p[0], p[1]), fmaxf(p[64], p[65]));
    lds[ic][ww] = (bf16_t)m;
  }
  __syncthreads();
  for (int i = threadIdx.x; i < 4096; i += 256) {
    int ww = i >> 7, ic = i & 127;
    pt[((size_t)(b * NKV) + hh * 32 + ww) * ICDIM + ic] = lds[ic][ww];
  }
}

// ---------------------------------------------------------------------------
// Fused flash-style attention, transposed formulation.
// sT[kv,q] = phiT(A) x theta(B);  yT[ic,q] = gT(A) x expT(B);  y /= sumexp.
// grid (NQ/16, B), block 32 (one wave per 16-query tile).
// ---------------------------------------------------------------------------
__global__ void __launch_bounds__(32) k_attn(const bf16_t* __restrict__ theta,
                                             const bf16_t* __restrict__ phiT,
                                             const bf16_t* __restrict__ gT,
                                             bf16_t* __restrict__ y) {
  int l = threadIdx.x, lm = l & 15, hi = l >> 4;
  int qt = blockIdx.x * 16, b = blockIdx.y;
  const bf16_t* th = theta + (size_t)b * NQ * ICDIM;
  const bf16_t* ph = phiT + (size_t)b * NKV * ICDIM;
  const bf16_t* gg = gT + (size_t)b * ICDIM * NKV;

  // preload theta B-operands (K = ic, 4 chunks of 32)
  BF16x16 Bth[4];
#pragma unroll
  for (int kk = 0; kk < 4; ++kk) {
    const bf16_t* p = th + (size_t)(qt + lm) * ICDIM + kk * 32 + hi * 16;
    Bth[kk].q[0] = *(const uint4*)(p);
    Bth[kk].q[1] = *(const uint4*)(p + 8);
  }

  v8f acc[8];
#pragma unroll
  for (int i = 0; i < 8; ++i) acc[i] = v8zero();
  v8f ssum = v8zero();

  for (int kv = 0; kv < NKV; kv += 32) {
    v8f s0 = v8zero(), s1 = v8zero();
#pragma unroll
    for (int kk = 0; kk < 4; ++kk) {
      BF16x16 A0, A1;
      const bf16_t* p0 = ph + (size_t)(kv + lm) * ICDIM + kk * 32 + hi * 16;
      A0.q[0] = *(const uint4*)(p0);
      A0.q[1] = *(const uint4*)(p0 + 8);
      const bf16_t* p1 = p0 + (size_t)16 * ICDIM;
      A1.q[0] = *(const uint4*)(p1);
      A1.q[1] = *(const uint4*)(p1 + 8);
      s0 = wmma_bf16(A0.v, Bth[kk].v, s0);
      s1 = wmma_bf16(A1.v, Bth[kk].v, s1);
    }
#pragma unroll
    for (int j = 0; j < 8; ++j) {
      s0[j] = __expf(s0[j]);
      s1[j] = __expf(s1[j]);
    }
    ssum += s0;
    ssum += s1;

    // repack exp'd C-layout tiles into a 32x16 bf16 B-operand (K = kv)
    BF16x16 Bp;
#pragma unroll
    for (int r = 0; r < 4; ++r) {
      unsigned int pa = pk2(s0[2 * r], s0[2 * r + 1]);      // kv rows 2r,2r+1 (lanes 0-15)
      unsigned int pb = pk2(s1[2 * r], s1[2 * r + 1]);      // kv rows 16+2r (lanes 0-15)
      unsigned int pbx = __shfl_xor(pb, 16, 32);
      Bp.u[r] = hi ? pbx : pa;
    }
#pragma unroll
    for (int r = 4; r < 8; ++r) {
      unsigned int pa = pk2(s0[2 * r - 8], s0[2 * r - 7]);  // kv rows 2r (lanes 16-31)
      unsigned int pax = __shfl_xor(pa, 16, 32);
      unsigned int pb = pk2(s1[2 * r - 8], s1[2 * r - 7]);  // kv rows 16+2r (lanes 16-31)
      Bp.u[r] = hi ? pb : pax;
    }

#pragma unroll
    for (int it = 0; it < 8; ++it) {
      BF16x16 Ag;
      const bf16_t* pg = gg + (size_t)(it * 16 + lm) * NKV + kv + hi * 16;
      Ag.q[0] = *(const uint4*)(pg);
      Ag.q[1] = *(const uint4*)(pg + 8);
      acc[it] = wmma_bf16(Ag.v, Bp.v, acc[it]);
    }
  }

  // softmax denominator: lane-local partial + half-wave swap
  float tot = 0.f;
#pragma unroll
  for (int j = 0; j < 8; ++j) tot += ssum[j];
  tot += __shfl_xor(tot, 16, 32);
  float inv = 1.0f / tot;

#pragma unroll
  for (int it = 0; it < 8; ++it) {
    BF16x8 o8;
#pragma unroll
    for (int j = 0; j < 8; ++j) o8.h[j] = (bf16_t)(acc[it][j] * inv);
    *(uint4*)(y + ((size_t)b * NQ + qt + lm) * ICDIM + it * 16 + hi * 8) = o8.q;
  }
}

// ---------------------------------------------------------------------------
// BN training-mode stats: per channel o, mean + rsqrt(var+eps) over B*NQ
// ---------------------------------------------------------------------------
__global__ void k_bn(const float* __restrict__ Wy, float* __restrict__ st) {
  __shared__ float sh[256], sh2[256];
  int o = blockIdx.x, t = threadIdx.x;
  float s = 0.f, s2 = 0.f;
  for (int i = t; i < BATCH * NQ; i += 256) {
    int b = i >> 12, q = i & (NQ - 1);
    float v = Wy[((size_t)b * CDIM + o) * NQ + q];
    s += v;
    s2 += v * v;
  }
  sh[t] = s;
  sh2[t] = s2;
  __syncthreads();
  for (int w = 128; w > 0; w >>= 1) {
    if (t < w) { sh[t] += sh[t + w]; sh2[t] += sh2[t + w]; }
    __syncthreads();
  }
  if (t == 0) {
    const float n = (float)(BATCH * NQ);
    float mean = sh[0] / n;
    float var = sh2[0] / n - mean * mean;
    st[o * 2] = mean;
    st[o * 2 + 1] = rsqrtf(var + 1e-5f);
  }
}

// ---------------------------------------------------------------------------
// BN apply + residual + 2x nearest upsample. one thread per pooled element.
// ---------------------------------------------------------------------------
__global__ void k_final(const float* __restrict__ Wy, const float* __restrict__ st,
                        const float* __restrict__ gamma, const float* __restrict__ beta,
                        const float* __restrict__ rgbp, float* __restrict__ out) {
  int i = blockIdx.x * 256 + threadIdx.x;  // < 8*256*4096
  int q = i & 4095, o = (i >> 12) & 255, b = i >> 20;
  int h = q >> 6, w = q & 63;
  float v = Wy[i];
  v = (v - st[o * 2]) * st[o * 2 + 1] * gamma[o] + beta[o] + rgbp[i];
  size_t base = (((size_t)(b * CDIM + o) * 128) + 2 * h) * 128 + 2 * w;
  float2 vv = {v, v};
  *(float2*)(out + base) = vv;
  *(float2*)(out + base + 128) = vv;
}

// ---------------------------------------------------------------------------
extern "C" void kernel_launch(void* const* d_in, const int* in_sizes, int n_in,
                              void* d_out, int out_size, void* d_ws, size_t ws_size,
                              hipStream_t stream) {
  const float* rgb   = (const float*)d_in[0];
  const float* event = (const float*)d_in[1];
  const float* g_w   = (const float*)d_in[2];
  const float* g_b   = (const float*)d_in[3];
  const float* th_w  = (const float*)d_in[4];
  const float* th_b  = (const float*)d_in[5];
  const float* ph_w  = (const float*)d_in[6];
  const float* ph_b  = (const float*)d_in[7];
  const float* W_w   = (const float*)d_in[8];
  const float* W_b   = (const float*)d_in[9];
  const float* gam   = (const float*)d_in[10];
  const float* bet   = (const float*)d_in[11];
  float* out = (float*)d_out;

  char* ws = (char*)d_ws;
  float*  rgbp   = (float*)(ws + 0);            // 32 MiB  f32 [b][c][4096]
  bf16_t* rgbT   = (bf16_t*)(ws + 33554432u);   // 16 MiB  bf16 [b][q][256]
  bf16_t* evT    = (bf16_t*)(ws + 50331648u);   // 16 MiB  bf16 [b][q][256]
  float*  gc     = (float*)(ws + 67108864u);    // 16 MiB  f32 [b][ic][4096]
  float*  phic   = (float*)(ws + 83886080u);    // 16 MiB  f32 [b][ic][4096]
  float*  Wy     = (float*)(ws + 67108864u);    // 32 MiB  f32 [b][o][4096] (aliases gc+phic)
  bf16_t* thetaB = (bf16_t*)(ws + 100663296u);  // 8 MiB   bf16 [b][q][128]
  bf16_t* gTb    = (bf16_t*)(ws + 109051904u);  // 2 MiB   bf16 [b][ic][1024]
  bf16_t* phiTb  = (bf16_t*)(ws + 111149056u);  // 2 MiB   bf16 [b][kv][128]
  bf16_t* yB     = (bf16_t*)(ws + 113246208u);  // 8 MiB   bf16 [b][q][128]
  bf16_t* wbf    = (bf16_t*)(ws + 121634816u);  // 256 KiB bf16 weights
  float*  bnst   = (float*)(ws + 121896960u);   // 2 KiB   f32 [256][2]

  k_pool_in<true><<<dim3(4, 64, 8), 256, 0, stream>>>(rgb, rgbp, rgbT);
  k_pool_in<false><<<dim3(4, 64, 8), 256, 0, stream>>>(event, nullptr, evT);
  k_cvt_w<<<512, 256, 0, stream>>>(th_w, g_w, ph_w, W_w, wbf);

  k_gemm_qt<<<dim3(256, 8, 8), 32, 0, stream>>>(rgbT, wbf + 0, th_b, thetaB, CDIM, ICDIM);
  k_gemm_ct<<<dim3(256, 8, 8), 32, 0, stream>>>(evT, wbf + 32768, g_b, gc, CDIM, ICDIM);
  k_gemm_ct<<<dim3(256, 8, 8), 32, 0, stream>>>(evT, wbf + 65536, ph_b, phic, CDIM, ICDIM);

  k_pool_g<<<4096, 256, 0, stream>>>(gc, gTb);
  k_pool_phi<<<dim3(32, 8), 256, 0, stream>>>(phic, phiTb);

  k_attn<<<dim3(256, 8), 32, 0, stream>>>(thetaB, phiTb, gTb, yB);

  k_gemm_ct<<<dim3(256, 16, 8), 32, 0, stream>>>(yB, wbf + 98304, W_b, Wy, ICDIM, CDIM);

  k_bn<<<256, 256, 0, stream>>>(Wy, bnst);
  k_final<<<32768, 256, 0, stream>>>(Wy, bnst, gam, bet, rgbp, out);
}
